// SpikeNet_89687507076361
// MI455X (gfx1250) — compile-verified, matrix operand order checked by reading, not run on hardware
//
#include <hip/hip_runtime.h>
#include <hip/hip_bf16.h>

// ---------------------------------------------------------------------------
// SpikeNet fused forward for MI455X (gfx1250, wave32, WMMA).
// TAU==1.0 collapses the LIF recurrence (v <- cur): the 8 timesteps are
// independent. Per block of 16 nodes we fuse gather -> L0 GEMM -> spike ->
// L1 GEMM -> spike -> pool GEMM, accumulating the pooled output in registers
// across t (zero intermediate global traffic).
// Roofline: ~1.23 GB of random 512B row gathers dominates (~53us @ 23.3TB/s);
// WMMA compute (69 GFLOP, f16 in / f32 acc) is far below that budget. The
// gathers for A-tile m+1 are issued BEFORE the WMMA chain for tile m
// (ping-pong LDS buffers, one barrier per tile). Crucially, only the S2=2
// gather path (6 float4 = 24 VGPRs) is ever live across the WMMA section:
// the S1=5 path exists only for tile 0 and is committed in the prologue.
// This keeps total pressure (64 resident B-frags + payload) below spill.
// ---------------------------------------------------------------------------

typedef __attribute__((ext_vector_type(16))) _Float16 v16h;
typedef __attribute__((ext_vector_type(8)))  _Float16 v8h;
typedef __attribute__((ext_vector_type(8)))  float    v8f;

#define NB      500000
#define FDIM    128
#define BNODES  20000
#define TSTEPS  8
#define H1DIM   128
#define H2DIM   64
#define CDIM    32
#define S1N     5
#define S2N     2
#define BT      16      // nodes per block
#define NTHREADS 256    // 8 wave32

// d_ws layout (f16 elements unless noted):
//   [0      , 32768)  W0T : (n=128, k=256)  [w_l0 ; w_r0] transposed
//   [32768  , 49152)  W1T : (n=64 , k=256)  [w_l1 ; w_r1] transposed
//   [49152  , 65536)  WPT : (n=32 , k=512)  w_pool transposed
//   then 192 floats: bias0[128] = b_l0+b_r0, bias1[64] = b_l1+b_r1
#define WS_F16_ELEMS 65536

// --------------------------- weight prep kernel ----------------------------
__global__ void spikenet_prep(const float* __restrict__ w_l0,
                              const float* __restrict__ w_r0,
                              const float* __restrict__ w_l1,
                              const float* __restrict__ w_r1,
                              const float* __restrict__ w_pool,
                              const float* __restrict__ b_l0,
                              const float* __restrict__ b_r0,
                              const float* __restrict__ b_l1,
                              const float* __restrict__ b_r1,
                              _Float16* __restrict__ ws16,
                              float* __restrict__ biases) {
  int i = blockIdx.x * blockDim.x + threadIdx.x;   // 0 .. 65535
  if (i < 32768) {                                  // W0T
    int n = i >> 8, k = i & 255;
    float v = (k < 128) ? w_l0[k * 128 + n] : w_r0[(k - 128) * 128 + n];
    ws16[n * 256 + k] = (_Float16)v;
  } else if (i < 49152) {                           // W1T
    int j = i - 32768; int n = j >> 8, k = j & 255;
    float v = (k < 128) ? w_l1[k * 64 + n] : w_r1[(k - 128) * 64 + n];
    ws16[32768 + n * 256 + k] = (_Float16)v;
  } else {                                          // WPT
    int j = i - 49152; int n = j >> 9, k = j & 511;
    ws16[49152 + n * 512 + k] = (_Float16)w_pool[k * 32 + n];
  }
  if (i < 128) biases[i] = b_l0[i] + b_r0[i];
  if (i >= 128 && i < 192) biases[i] = b_l1[i - 128] + b_r1[i - 128];
}

// Per-lane WMMA fragment load (16-bit A/B, 16x16x32):
// lanes 0-15 hold K chunks [0..7] and [16..23]; lanes 16-31 hold [8..15] and
// [24..31] (ISA 7.12.2). rowbase points at one matrix row (A: M-row from LDS,
// B: N-column from the transposed weight copy). Both chunks are 16B loads.
__device__ __forceinline__ v16h load_frag(const _Float16* rowbase, int hi8, int koff) {
  v8h lo = *(const v8h*)(rowbase + koff + hi8);
  v8h hi = *(const v8h*)(rowbase + koff + 16 + hi8);
  v16h r;
#pragma unroll
  for (int i = 0; i < 8; ++i) { r[i] = lo[i]; r[8 + i] = hi[i]; }
  return r;
}

// ------------------------------ main kernel --------------------------------
__global__ void __launch_bounds__(NTHREADS)
spikenet_fused(const float* __restrict__ x,
               const int*   __restrict__ nodes,
               const int*   __restrict__ nbr1,
               const int*   __restrict__ nbr2,
               const _Float16* __restrict__ ws16,
               const float* __restrict__ biases,
               const float* __restrict__ b_pool,
               float* __restrict__ out) {
  __shared__ _Float16 Abuf0[16 * 256];   // ping A-tile (self||neigh), 8 KB
  __shared__ _Float16 Abuf1[16 * 256];   // pong A-tile, 8 KB
  __shared__ _Float16 S0[96 * 128];      // layer-0 spikes for 16 nodes, 24 KB
  __shared__ _Float16 S1b[16 * 64];      // layer-1 spikes, 2 KB
  // Abuf0's first 16*128 is reused as the layer-1 neighbor-mean tile.

  const _Float16* W0T = ws16;            // (128,256)
  const _Float16* W1T = ws16 + 32768;    // (64,256)
  const _Float16* WPT = ws16 + 49152;    // (32,512)

  const int tid  = threadIdx.x;
  const int lane = tid & 31;
  const int wave = tid >> 5;
  const int hi8  = (lane >> 4) << 3;     // 0 or 8 : K-chunk select per lane half
  const int row16 = lane & 15;
  const int b0 = blockIdx.x * BT;
  const int ncol = (wave << 4) + row16;  // this wave's output column (layer 0)

  const int r  = tid >> 4;               // staging: row within tile (0..15)
  const int c  = (tid & 15) << 3;        // staging: 8 columns per thread
  // Pipelined tiles 1..5 always take the nbr1/nbr2 path; precompute per-row
  // quantities that do not depend on m:  R = m*16 + r, q = R-16.
  v8f pacc = {};                         // pooled output accumulator (waves 0,1)

  for (int t = 0; t < TSTEPS; ++t) {
    const int* nb1t = nbr1 + (size_t)t * (BNODES * S1N);
    const int* nb2t = nbr2 + (size_t)t * (BNODES * S1N * S2N);

    // ---- layer 0: 96 rows = 6 M-tiles of 16, K=256, N=128 (8 waves x 16) --
    // Prologue: tile 0 (the only S1=5 tile) gathered and committed at once;
    // its 12-float4 payload is never live across a WMMA section.
    {
      int b = b0 + r;
      const float* xs = x + (size_t)nodes[b] * FDIM + c;
      float4 s0v = *(const float4*)(xs);
      float4 s1v = *(const float4*)(xs + 4);
      const int* nb = nb1t + b * S1N;
      float acc[8] = {0.f,0.f,0.f,0.f,0.f,0.f,0.f,0.f};
      for (int s = 0; s < S1N; ++s) {
        const float* xn = x + (size_t)nb[s] * FDIM + c;
        float4 a = *(const float4*)(xn);
        float4 bb = *(const float4*)(xn + 4);
        acc[0] += a.x;  acc[1] += a.y;  acc[2] += a.z;  acc[3] += a.w;
        acc[4] += bb.x; acc[5] += bb.y; acc[6] += bb.z; acc[7] += bb.w;
      }
      _Float16* dst = Abuf0 + r * 256 + c;
      dst[0] = (_Float16)s0v.x; dst[1] = (_Float16)s0v.y;
      dst[2] = (_Float16)s0v.z; dst[3] = (_Float16)s0v.w;
      dst[4] = (_Float16)s1v.x; dst[5] = (_Float16)s1v.y;
      dst[6] = (_Float16)s1v.z; dst[7] = (_Float16)s1v.w;
      _Float16* dstn = dst + 128;
#pragma unroll
      for (int i = 0; i < 8; ++i) dstn[i] = (_Float16)(acc[i] * 0.2f);
    }
    __syncthreads();

    // Steady state: prefetch tile m+1 (always S2=2: 6 float4 payload) before
    // running tile m's WMMA chain; commit after; one barrier per tile.
    float4 g0, g1, g2, g3, g4, g5;       // self row + 2 neighbor rows
    for (int m = 0; m < 6; ++m) {
      const _Float16* cur = (m & 1) ? Abuf1 : Abuf0;
      _Float16*       nxt = (m & 1) ? Abuf0 : Abuf1;
      if (m < 5) {   // fire next tile's global gathers NOW (no wait yet)
        int q  = m * 16 + r;             // (m+1)*16 + r - 16
        int bp = q / 5, s = q % 5;
        int j1 = (b0 + bp) * S1N + s;
        const float* xs = x + (size_t)nb1t[j1] * FDIM + c;
        const int* nb = nb2t + j1 * S2N;
        g0 = *(const float4*)(xs);
        g1 = *(const float4*)(xs + 4);
        const float* xn0 = x + (size_t)nb[0] * FDIM + c;
        const float* xn1 = x + (size_t)nb[1] * FDIM + c;
        g2 = *(const float4*)(xn0);
        g3 = *(const float4*)(xn0 + 4);
        g4 = *(const float4*)(xn1);
        g5 = *(const float4*)(xn1 + 4);
      }

      { // WMMA: wave w owns output columns [w*16, w*16+16)
        v8f cacc = {};
#pragma unroll
        for (int k = 0; k < 8; ++k) {
          v16h af = load_frag(cur + row16 * 256, hi8, k * 32);
          v16h bf = load_frag(W0T + ncol * 256, hi8, k * 32); // reg-resident
          cacc = __builtin_amdgcn_wmma_f32_16x16x32_f16(
                     false, af, false, bf, (short)0, cacc, false, false);
        }
        float bias = biases[ncol];
        int rbase = m * 16 + ((lane >> 4) << 3);
#pragma unroll
        for (int j = 0; j < 8; ++j) {
          float v = cacc[j] + bias;   // v0 = cur (TAU==1); spike at THRESH=1
          S0[(rbase + j) * 128 + ncol] = (v >= 1.0f) ? (_Float16)1.0f : (_Float16)0.0f;
        }
      }

      if (m < 5) {   // first use of the in-flight gathers: waits land here
        _Float16* dst = nxt + r * 256 + c;
        dst[0] = (_Float16)g0.x; dst[1] = (_Float16)g0.y;
        dst[2] = (_Float16)g0.z; dst[3] = (_Float16)g0.w;
        dst[4] = (_Float16)g1.x; dst[5] = (_Float16)g1.y;
        dst[6] = (_Float16)g1.z; dst[7] = (_Float16)g1.w;
        _Float16* dstn = dst + 128;
        dstn[0] = (_Float16)((g2.x + g4.x) * 0.5f);
        dstn[1] = (_Float16)((g2.y + g4.y) * 0.5f);
        dstn[2] = (_Float16)((g2.z + g4.z) * 0.5f);
        dstn[3] = (_Float16)((g2.w + g4.w) * 0.5f);
        dstn[4] = (_Float16)((g3.x + g5.x) * 0.5f);
        dstn[5] = (_Float16)((g3.y + g5.y) * 0.5f);
        dstn[6] = (_Float16)((g3.z + g5.z) * 0.5f);
        dstn[7] = (_Float16)((g3.w + g5.w) * 0.5f);
      }
      __syncthreads();
    }

    // ---- layer 1: M=16, K=256 ([h0b | mean_s s0_nbr]), N=64 --------------
    { // stage only the neighbor-mean half into Abuf0[0:16*128); the self half
      // (h0b = S0 rows 0..15) is read directly from S0 by the WMMA below.
      float acc[8] = {0.f,0.f,0.f,0.f,0.f,0.f,0.f,0.f};
      for (int s = 0; s < S1N; ++s) {
        const _Float16* src = S0 + (16 + r * S1N + s) * 128 + c;
#pragma unroll
        for (int i = 0; i < 8; ++i) acc[i] += (float)src[i];
      }
      _Float16* dstn = Abuf0 + r * 128 + c;
#pragma unroll
      for (int i = 0; i < 8; ++i) dstn[i] = (_Float16)(acc[i] * 0.2f);
    }
    __syncthreads();
    if (wave < 4) {            // 4 waves x 16 cols = N=64 (full EXEC per wave)
      v8f cacc = {};
#pragma unroll
      for (int k = 0; k < 4; ++k) {    // K 0..127: self spikes straight from S0
        v16h af = load_frag(S0 + row16 * 128, hi8, k * 32);
        v16h bf = load_frag(W1T + ncol * 256, hi8, k * 32);
        cacc = __builtin_amdgcn_wmma_f32_16x16x32_f16(
                   false, af, false, bf, (short)0, cacc, false, false);
      }
#pragma unroll
      for (int k = 0; k < 4; ++k) {    // K 128..255: staged neighbor means
        v16h af = load_frag(Abuf0 + row16 * 128, hi8, k * 32);
        v16h bf = load_frag(W1T + ncol * 256, hi8, (4 + k) * 32);
        cacc = __builtin_amdgcn_wmma_f32_16x16x32_f16(
                   false, af, false, bf, (short)0, cacc, false, false);
      }
      float bias = biases[128 + ncol];
      int rbase = (lane >> 4) << 3;
#pragma unroll
      for (int j = 0; j < 8; ++j) {
        float v = cacc[j] + bias;
        S1b[(rbase + j) * 64 + ncol] = (v >= 1.0f) ? (_Float16)1.0f : (_Float16)0.0f;
      }
    }
    __syncthreads();
    // ---- pool: out += s1_t @ w_pool[t*64:(t+1)*64]; K=64, N=32 (waves 0,1)
    if (wave < 2) {
      v16h af0 = load_frag(S1b + row16 * 64, hi8, 0);
      v16h af1 = load_frag(S1b + row16 * 64, hi8, 32);
      v16h bf0 = load_frag(WPT + ncol * 512 + t * 64, hi8, 0);
      v16h bf1 = load_frag(WPT + ncol * 512 + t * 64, hi8, 32);
      pacc = __builtin_amdgcn_wmma_f32_16x16x32_f16(
                 false, af0, false, bf0, (short)0, pacc, false, false);
      pacc = __builtin_amdgcn_wmma_f32_16x16x32_f16(
                 false, af1, false, bf1, (short)0, pacc, false, false);
    }
    __syncthreads();   // S1b / Abuf0 reuse next timestep
  }

  if (wave < 2) {     // write out (B,32): 2 waves x 16 cols, 16 rows
    float bp = b_pool[ncol];
    int rbase = (lane >> 4) << 3;
#pragma unroll
    for (int j = 0; j < 8; ++j) {
      out[(size_t)(b0 + rbase + j) * CDIM + ncol] = pacc[j] + bp;
    }
  }
}

// ------------------------------- launcher ----------------------------------
extern "C" void kernel_launch(void* const* d_in, const int* in_sizes, int n_in,
                              void* d_out, int out_size, void* d_ws, size_t ws_size,
                              hipStream_t stream) {
  const float* x      = (const float*)d_in[0];
  const int*   nodes  = (const int*)  d_in[1];
  const int*   nbr1   = (const int*)  d_in[2];
  const int*   nbr2   = (const int*)  d_in[3];
  const float* w_l0   = (const float*)d_in[4];
  const float* b_l0   = (const float*)d_in[5];
  const float* w_r0   = (const float*)d_in[6];
  const float* b_r0   = (const float*)d_in[7];
  const float* w_l1   = (const float*)d_in[8];
  const float* b_l1   = (const float*)d_in[9];
  const float* w_r1   = (const float*)d_in[10];
  const float* b_r1   = (const float*)d_in[11];
  const float* w_pool = (const float*)d_in[12];
  const float* b_pool = (const float*)d_in[13];
  float* out = (float*)d_out;

  _Float16* ws16 = (_Float16*)d_ws;
  float* biases  = (float*)((char*)d_ws + (size_t)WS_F16_ELEMS * sizeof(_Float16));

  // Stage f16 transposed weights + fused biases into scratch (L2-resident).
  spikenet_prep<<<WS_F16_ELEMS / NTHREADS, NTHREADS, 0, stream>>>(
      w_l0, w_r0, w_l1, w_r1, w_pool, b_l0, b_r0, b_l1, b_r1, ws16, biases);

  // 20000 nodes / 16 per block = 1250 blocks, 8 wave32 each.
  spikenet_fused<<<BNODES / BT, NTHREADS, 0, stream>>>(
      x, nodes, nbr1, nbr2, ws16, biases, b_pool, out);
}
